// MultiHeadAttention_23828478558304
// MI455X (gfx1250) — matrix operand order repaired and checked by reading, not compiled, not run on hardware
//
#include <hip/hip_runtime.h>

#define DM  1024
#define NH  16
#define DH  64
#define BB  4
#define SS  2048
#define TOK (BB * SS)

typedef __attribute__((ext_vector_type(16))) __bf16        v16bf;
typedef __attribute__((ext_vector_type(8)))  float         v8f;
typedef __attribute__((ext_vector_type(4)))  unsigned int  v4u;

union BF16x16 {
    v16bf v;
    v4u q[2];
    unsigned int u32[8];
};

static __device__ __forceinline__ unsigned short f32_to_bf16(float f) {
    unsigned int u = __float_as_uint(f);
    unsigned int r = 0x7FFFu + ((u >> 16) & 1u);
    return (unsigned short)((u + r) >> 16);
}

// CDNA5 async global->LDS copy (ASYNCcnt-tracked, CDNA5 ISA)
static __device__ __forceinline__ void async_ld_b128(unsigned lds_byte_addr,
                                                     const void* src) {
    asm volatile("global_load_async_to_lds_b128 %0, %1, off"
                 :: "v"(lds_byte_addr), "v"(src)
                 : "memory");
}
#define WAIT_ASYNCCNT(n) asm volatile("s_wait_asynccnt " #n ::: "memory")

// Raw split workgroup barrier WITHOUT the acq_rel VMEM drain of __syncthreads.
// LDS visibility of async-staged tiles is ordered by s_wait_asynccnt before the
// signal; DS reads are consumed (dscnt-waited) by WMMAs before the signal.
static __device__ __forceinline__ void wg_barrier() {
    asm volatile("s_barrier_signal -1\n\ts_barrier_wait -1" ::: "memory");
}

static __device__ __forceinline__ unsigned lds_addr_of(const void* p) {
    // low 32 bits of a flat shared address are the LDS byte address (ISA 10.2)
    return (unsigned)(uintptr_t)p;
}

// ---------------- fp32 -> bf16 conversion ----------------
__global__ void cvt_f32_bf16(const float* __restrict__ in,
                             unsigned short* __restrict__ out, size_t n) {
    size_t i = (size_t)blockIdx.x * blockDim.x + threadIdx.x;
    size_t stride = (size_t)gridDim.x * blockDim.x;
    for (; i < n; i += stride) out[i] = f32_to_bf16(in[i]);
}

// ---------------- WMMA GEMM: out = A[M,K] * Wt[N,K]^T + bias ----------------
// mode 0: f32 out [M,N]
// mode 1: bf16 out [B,H,S,DH]   (Q,K head-major)
// mode 2: bf16 out [B,H,DH,S]   (V transposed for contiguous P*V B-loads)
__global__ __launch_bounds__(256) void gemm_bf16_wmma(
    const unsigned short* __restrict__ A,
    const unsigned short* __restrict__ Wt,
    const float* __restrict__ bias,
    void* __restrict__ out,
    int M, int N, int K, int mode, int Sdim)
{
    __shared__ __align__(16) unsigned short ldsB[2][64][40];  // rows padded: 80B

    const int tid  = threadIdx.x;
    const int lane = tid & 31;
    const int wv   = tid >> 5;
    const int kh   = lane >> 4;
    const int ln   = lane & 15;
    const int row0 = blockIdx.x * 128 + wv * 16;
    const int col0 = blockIdx.y * 64;

    const unsigned short* arow = A + (size_t)(row0 + ln) * K;
    const int scol = tid >> 2;          // staging: column this thread copies
    const int schk = tid & 3;           // 16B chunk within the 64B k-run
    const unsigned short* wsrc = Wt + (size_t)(col0 + scol) * K + schk * 8;
    const unsigned ldsB0 = lds_addr_of(&ldsB[0][0][0]) + scol * 80 + schk * 16;
    const unsigned ldsB1 = lds_addr_of(&ldsB[1][0][0]) + scol * 80 + schk * 16;

    v8f acc[4];
    #pragma unroll
    for (int t = 0; t < 4; ++t)
        #pragma unroll
        for (int r = 0; r < 8; ++r) acc[t][r] = 0.0f;

    auto loadA = [&](BF16x16& a, int k0) {
        a.q[0] = *(const v4u*)(arow + k0 + kh * 16);
        a.q[1] = *(const v4u*)(arow + k0 + kh * 16 + 8);
    };
    auto compute = [&](const BF16x16& a, const unsigned short (*lb)[40]) {
        BF16x16 b[4];
        #pragma unroll
        for (int t = 0; t < 4; ++t) {
            const unsigned short* bp = &lb[t * 16 + ln][kh * 16];
            b[t].q[0] = ((const v4u*)bp)[0];
            b[t].q[1] = ((const v4u*)bp)[1];
        }
        #pragma unroll
        for (int t = 0; t < 4; ++t)
            acc[t] = __builtin_amdgcn_wmma_f32_16x16x32_bf16(
                false, a.v, false, b[t].v, (short)0, acc[t], false, false);
    };

    BF16x16 a0, a1;
    loadA(a0, 0);
    async_ld_b128(ldsB0, wsrc);                         // stage k=0 -> buf0
    for (int k0 = 0; k0 + 64 <= K; k0 += 64) {
        // prefetch k0+32 -> buf1 while computing k0 from buf0
        loadA(a1, k0 + 32);
        async_ld_b128(ldsB1, wsrc + k0 + 32);
        WAIT_ASYNCCNT(1);
        wg_barrier();
        compute(a0, ldsB[0]);
        wg_barrier();
        // prefetch k0+64 -> buf0 while computing k0+32 from buf1
        if (k0 + 64 < K) {
            loadA(a0, k0 + 64);
            async_ld_b128(ldsB0, wsrc + k0 + 64);
            WAIT_ASYNCCNT(1);
        } else {
            WAIT_ASYNCCNT(0);
        }
        wg_barrier();
        compute(a1, ldsB[1]);
        wg_barrier();
    }

    // Epilogue: D layout row = r + 8*kh, col = col0 + t*16 + ln
    #pragma unroll
    for (int t = 0; t < 4; ++t) {
        const int n = col0 + t * 16 + ln;
        const float bvv = bias[n];
        #pragma unroll
        for (int r = 0; r < 8; ++r) {
            const int m = row0 + r + 8 * kh;
            float val = acc[t][r] + bvv;
            if (mode == 0) {
                ((float*)out)[(size_t)m * N + n] = val;
            } else {
                int bidx = m / Sdim, s = m - bidx * Sdim;
                int h = n >> 6, dh = n & 63;
                unsigned short bf = f32_to_bf16(val);
                if (mode == 1)
                    ((unsigned short*)out)[(((size_t)bidx * NH + h) * Sdim + s) * DH + dh] = bf;
                else
                    ((unsigned short*)out)[(((size_t)bidx * NH + h) * DH + dh) * Sdim + s] = bf;
            }
        }
    }
}

// ---------------- flash attention (one wave = 16 query rows) ----------------
__global__ __launch_bounds__(256) void flash_attn_wmma(
    const unsigned short* __restrict__ Qh,   // [B*H, S, DH] bf16
    const unsigned short* __restrict__ Kh,   // [B*H, S, DH] bf16
    const unsigned short* __restrict__ Vt,   // [B*H, DH, S] bf16
    unsigned short* __restrict__ ctx,        // [B, S, DM] bf16
    int S)
{
    __shared__ __align__(16) unsigned short ldsK[2][32][72];  // rows padded: 144B
    __shared__ __align__(16) unsigned short ldsV[2][64][40];  // rows padded: 80B
    __shared__ __align__(16) unsigned short ldsP[8][16][34];  // per-wave P tile

    const int tid  = threadIdx.x;
    const int lane = tid & 31;
    const int wv   = tid >> 5;
    const int kh   = lane >> 4;
    const int ln   = lane & 15;
    const int bh   = blockIdx.x;
    const int b    = bh >> 4;
    const int h    = bh & 15;
    const int q0   = blockIdx.y * 128 + wv * 16;

    const unsigned short* Qp = Qh + (size_t)bh * S * DH;
    const unsigned short* Kp = Kh + (size_t)bh * S * DH;
    const unsigned short* Vp = Vt + (size_t)bh * DH * S;

    // staging address precompute
    const int kr = tid >> 3, kc = tid & 7;   // K: 32 rows x 8 chunks of 16B
    const int vr = tid >> 2, vc = tid & 3;   // V: 64 rows x 4 chunks of 16B
    const unsigned short* ksrc = Kp + (size_t)kr * DH + kc * 8;
    const unsigned short* vsrc = Vp + (size_t)vr * S + vc * 8;
    const unsigned ldsK_d[2] = {
        lds_addr_of(&ldsK[0][0][0]) + kr * 144 + kc * 16,
        lds_addr_of(&ldsK[1][0][0]) + kr * 144 + kc * 16 };
    const unsigned ldsV_d[2] = {
        lds_addr_of(&ldsV[0][0][0]) + vr * 80 + vc * 16,
        lds_addr_of(&ldsV[1][0][0]) + vr * 80 + vc * 16 };

    // Q A-tiles for the two 32-wide d chunks; resident all loop long
    BF16x16 qa[2];
    #pragma unroll
    for (int d = 0; d < 2; ++d) {
        const unsigned short* qp = Qp + (size_t)(q0 + ln) * DH + d * 32;
        qa[d].q[0] = *(const v4u*)(qp + kh * 16);
        qa[d].q[1] = *(const v4u*)(qp + kh * 16 + 8);
    }

    v8f acc[4];
    #pragma unroll
    for (int t = 0; t < 4; ++t)
        #pragma unroll
        for (int r = 0; r < 8; ++r) acc[t][r] = 0.0f;

    float mrow[8], lrow[8];
    #pragma unroll
    for (int r = 0; r < 8; ++r) { mrow[r] = -1e30f; lrow[r] = 0.0f; }

    unsigned short (*pl)[34] = ldsP[wv];
    const float scale = 0.125f;  // 1/sqrt(64)

    auto stage = [&](int j, int buf) {
        async_ld_b128(buf ? ldsK_d[1] : ldsK_d[0], ksrc + (size_t)j * DH);
        async_ld_b128(buf ? ldsV_d[1] : ldsV_d[0], vsrc + j);
    };

    auto attn_step = [&](const unsigned short (*lk)[72],
                         const unsigned short (*lv)[40]) {
        // ---- scores: two 16x16 D tiles over 32 keys ----
        v8f sc[2];
        #pragma unroll
        for (int t = 0; t < 2; ++t) {
            BF16x16 kb0, kb1;
            const unsigned short* kp = &lk[t * 16 + ln][0];
            kb0.q[0] = ((const v4u*)(kp + kh * 16))[0];
            kb0.q[1] = ((const v4u*)(kp + kh * 16))[1];
            kb1.q[0] = ((const v4u*)(kp + 32 + kh * 16))[0];
            kb1.q[1] = ((const v4u*)(kp + 32 + kh * 16))[1];
            v8f z = {0.f, 0.f, 0.f, 0.f, 0.f, 0.f, 0.f, 0.f};
            z = __builtin_amdgcn_wmma_f32_16x16x32_bf16(
                false, qa[0].v, false, kb0.v, (short)0, z, false, false);
            z = __builtin_amdgcn_wmma_f32_16x16x32_bf16(
                false, qa[1].v, false, kb1.v, (short)0, z, false, false);
            sc[t] = z;
        }

        // ---- online softmax (rows live in 16-lane halves) ----
        float bm[8];
        #pragma unroll
        for (int r = 0; r < 8; ++r) {
            float p0 = sc[0][r] * scale, p1 = sc[1][r] * scale;
            sc[0][r] = p0; sc[1][r] = p1;
            bm[r] = fmaxf(p0, p1);
        }
        #pragma unroll
        for (int off = 1; off < 16; off <<= 1)
            #pragma unroll
            for (int r = 0; r < 8; ++r)
                bm[r] = fmaxf(bm[r], __shfl_xor(bm[r], off, 32));

        float corr[8], rs[8];
        #pragma unroll
        for (int r = 0; r < 8; ++r) {
            float mn = fmaxf(mrow[r], bm[r]);
            corr[r] = __expf(mrow[r] - mn);
            mrow[r] = mn;
            float p0 = __expf(sc[0][r] - mn);
            float p1 = __expf(sc[1][r] - mn);
            sc[0][r] = p0; sc[1][r] = p1;
            rs[r] = p0 + p1;
        }
        #pragma unroll
        for (int off = 1; off < 16; off <<= 1)
            #pragma unroll
            for (int r = 0; r < 8; ++r)
                rs[r] += __shfl_xor(rs[r], off, 32);
        #pragma unroll
        for (int r = 0; r < 8; ++r) lrow[r] = lrow[r] * corr[r] + rs[r];
        #pragma unroll
        for (int t = 0; t < 4; ++t)
            #pragma unroll
            for (int r = 0; r < 8; ++r) acc[t][r] *= corr[r];

        // ---- P: D-layout regs -> wave-private LDS -> A-layout regs ----
        // (DS ops from one wave are processed in order; no barrier needed)
        #pragma unroll
        for (int t = 0; t < 2; ++t)
            #pragma unroll
            for (int r = 0; r < 8; ++r)
                pl[r + 8 * kh][t * 16 + ln] = f32_to_bf16(sc[t][r]);
        BF16x16 pa;
        #pragma unroll
        for (int v = 0; v < 8; ++v) {
            int k = kh * 16 + v * 2;
            pa.u32[v] = *(const unsigned int*)(&pl[ln][k]);
        }

        // ---- ctx += P * V ----
        #pragma unroll
        for (int dt = 0; dt < 4; ++dt) {
            BF16x16 vb;
            const unsigned short* vp = &lv[dt * 16 + ln][kh * 16];
            vb.q[0] = ((const v4u*)vp)[0];
            vb.q[1] = ((const v4u*)vp)[1];
            acc[dt] = __builtin_amdgcn_wmma_f32_16x16x32_bf16(
                false, pa.v, false, vb.v, (short)0, acc[dt], false, false);
        }
    };

    stage(0, 0);
    for (int j = 0; j + 64 <= S; j += 64) {
        stage(j + 32, 1);          // prefetch next block -> buf1
        WAIT_ASYNCCNT(2);          // buf0's 2 ops (older) complete
        wg_barrier();
        attn_step(ldsK[0], ldsV[0]);
        wg_barrier();
        if (j + 64 < S) {
            stage(j + 64, 0);
            WAIT_ASYNCCNT(2);
        } else {
            WAIT_ASYNCCNT(0);
        }
        wg_barrier();
        attn_step(ldsK[1], ldsV[1]);
        wg_barrier();
    }

    // ---- finalize and write ctx bf16 [B,S,DM] ----
    #pragma unroll
    for (int r = 0; r < 8; ++r) {
        float inv = 1.0f / lrow[r];
        int m = q0 + r + 8 * kh;
        size_t base = ((size_t)b * S + m) * DM + (size_t)h * DH;
        #pragma unroll
        for (int dt = 0; dt < 4; ++dt)
            ctx[base + dt * 16 + ln] = f32_to_bf16(acc[dt][r] * inv);
    }
}

extern "C" void kernel_launch(void* const* d_in, const int* in_sizes, int n_in,
                              void* d_out, int out_size, void* d_ws, size_t ws_size,
                              hipStream_t stream) {
    const float* q_in = (const float*)d_in[0];
    const float* k_in = (const float*)d_in[1];
    const float* v_in = (const float*)d_in[2];
    const float* Wq   = (const float*)d_in[3];
    const float* bq   = (const float*)d_in[4];
    const float* Wk   = (const float*)d_in[5];
    const float* bk   = (const float*)d_in[6];
    const float* Wv   = (const float*)d_in[7];
    const float* bv   = (const float*)d_in[8];
    const float* Wo   = (const float*)d_in[9];
    const float* bo   = (const float*)d_in[10];
    float* out = (float*)d_out;

    char* ws = (char*)d_ws;
    size_t off = 0;
    auto take = [&](size_t bytes) -> char* {
        char* p = ws + off;
        off += (bytes + 255) & ~(size_t)255;
        return p;
    };
    const size_t xbytes = (size_t)TOK * DM * sizeof(unsigned short);  // 16 MB
    const size_t wbytes = (size_t)DM * DM * sizeof(unsigned short);   // 2 MB
    unsigned short* xq  = (unsigned short*)take(xbytes);
    unsigned short* xk  = (unsigned short*)take(xbytes);
    unsigned short* xv  = (unsigned short*)take(xbytes);
    unsigned short* wqb = (unsigned short*)take(wbytes);
    unsigned short* wkb = (unsigned short*)take(wbytes);
    unsigned short* wvb = (unsigned short*)take(wbytes);
    unsigned short* wob = (unsigned short*)take(wbytes);
    unsigned short* Qh  = (unsigned short*)take(xbytes);
    unsigned short* Kh  = (unsigned short*)take(xbytes);
    unsigned short* Vt  = (unsigned short*)take(xbytes);
    unsigned short* ctxb = xq;  // xq is dead after the Q projection; reuse for ctx

    dim3 cb(256);
    cvt_f32_bf16<<<dim3(1024), cb, 0, stream>>>(q_in, xq, (size_t)TOK * DM);
    cvt_f32_bf16<<<dim3(1024), cb, 0, stream>>>(k_in, xk, (size_t)TOK * DM);
    cvt_f32_bf16<<<dim3(1024), cb, 0, stream>>>(v_in, xv, (size_t)TOK * DM);
    cvt_f32_bf16<<<dim3(256),  cb, 0, stream>>>(Wq, wqb, (size_t)DM * DM);
    cvt_f32_bf16<<<dim3(256),  cb, 0, stream>>>(Wk, wkb, (size_t)DM * DM);
    cvt_f32_bf16<<<dim3(256),  cb, 0, stream>>>(Wv, wvb, (size_t)DM * DM);
    cvt_f32_bf16<<<dim3(256),  cb, 0, stream>>>(Wo, wob, (size_t)DM * DM);

    dim3 ggrid(TOK / 128, DM / 64);
    gemm_bf16_wmma<<<ggrid, cb, 0, stream>>>(xq, wqb, bq, Qh, TOK, DM, DM, 1, SS);
    gemm_bf16_wmma<<<ggrid, cb, 0, stream>>>(xk, wkb, bk, Kh, TOK, DM, DM, 1, SS);
    gemm_bf16_wmma<<<ggrid, cb, 0, stream>>>(xv, wvb, bv, Vt, TOK, DM, DM, 2, SS);

    flash_attn_wmma<<<dim3(BB * NH, SS / 128), cb, 0, stream>>>(Qh, Kh, Vt, ctxb, SS);

    gemm_bf16_wmma<<<ggrid, cb, 0, stream>>>(ctxb, wob, bo, out, TOK, DM, DM, 0, SS);
}
